// Transform_48550310314489
// MI455X (gfx1250) — matrix operand order repaired and checked by reading, not compiled, not run on hardware
//
#include <hip/hip_runtime.h>
#include <hip/hip_bf16.h>
#include <stdint.h>

#define B_ 16
#define T_ 1024
#define C_ 256

typedef __attribute__((ext_vector_type(16))) __bf16 v16bf;
typedef __attribute__((ext_vector_type(16))) unsigned short u16x16;
typedef __attribute__((ext_vector_type(8)))  unsigned short u16x8;
typedef __attribute__((ext_vector_type(8)))  float v8f;
typedef __attribute__((ext_vector_type(4)))  unsigned int u32x4;
typedef __attribute__((ext_vector_type(8)))  int i32x8;
typedef __attribute__((ext_vector_type(4)))  int i32x4;

#ifdef __has_builtin
#if __has_builtin(__builtin_amdgcn_tensor_load_to_lds)
#define HAVE_TDM 1
#endif
#endif
#ifndef HAVE_TDM
#define HAVE_TDM 0
#endif

static __device__ __forceinline__ unsigned short f2bf(float f) {
  unsigned u = __builtin_bit_cast(unsigned, f);
  unsigned r = u + 0x7FFFu + ((u >> 16) & 1u);   // round-to-nearest-even
  return (unsigned short)(r >> 16);
}

static __device__ __forceinline__ v8f wmma_bf16(const u16x16& a, const u16x16& b, v8f c) {
  return __builtin_amdgcn_wmma_f32_16x16x32_bf16(
      false, __builtin_bit_cast(v16bf, a),
      false, __builtin_bit_cast(v16bf, b),
      (short)0, c, false, false);
}

static __device__ __forceinline__ u16x16 cat8(u16x8 l, u16x8 h) {
  return __builtin_shufflevector(l, h, 0,1,2,3,4,5,6,7,8,9,10,11,12,13,14,15);
}
static __device__ __forceinline__ u16x16 ld_afrag(const unsigned short* As, int mrow, int kh) {
  u16x8 lo = __builtin_bit_cast(u16x8, *(const uint4*)(As + mrow * 32 + kh * 8));
  u16x8 hi = __builtin_bit_cast(u16x8, *(const uint4*)(As + mrow * 32 + 16 + kh * 8));
  return cat8(lo, hi);
}
static __device__ __forceinline__ u16x16 ld_bfrag(const unsigned short* Bs, int n, int kh) {
  const uint4* p = (const uint4*)(Bs + n * 32 + kh * 8);
  return cat8(__builtin_bit_cast(u16x8, p[0]), __builtin_bit_cast(u16x8, p[2]));
}

// ---- CDNA5 async global->LDS (ASYNCcnt) ----
static __device__ __forceinline__ void async_b128(void* lds, const void* gmem) {
  unsigned off = (unsigned)(size_t)lds;   // low 32 bits of flat shared addr == LDS offset
  asm volatile("global_load_async_to_lds_b128 %0, %1, off"
               :: "v"(off), "v"(gmem) : "memory");
}
static __device__ __forceinline__ void async_b64(void* lds, const void* gmem) {
  unsigned off = (unsigned)(size_t)lds;
  asm volatile("global_load_async_to_lds_b64 %0, %1, off"
               :: "v"(off), "v"(gmem) : "memory");
}
template <int N>
static __device__ __forceinline__ void async_waitN() {
  asm volatile("s_wait_asynccnt %0" :: "i"(N) : "memory");
}
template <int N>
static __device__ __forceinline__ void tensor_waitN() {
  asm volatile("s_wait_tensorcnt %0" :: "i"(N) : "memory");
}

#if HAVE_TDM
// TDM 2D tile load: tile_dim0=32 elems (contiguous), tile_dim1=256 rows,
// tensor row length/stride 256 elems, 2-byte elements, linear LDS dest.
static __device__ __forceinline__ void tdm_load_2d(unsigned ldsByteAddr, const void* gaddr) {
  unsigned long long ga = (unsigned long long)(size_t)gaddr;
  u32x4 g0;
  g0[0] = 1u;                                           // count=1 valid descriptor
  g0[1] = ldsByteAddr;                                  // lds_addr
  g0[2] = (unsigned)(ga & 0xFFFFFFFFu);                 // global_addr[31:0]
  g0[3] = (unsigned)((ga >> 32) & 0x01FFFFFFu) | 0x80000000u;  // addr[56:32] | type=2
  i32x8 g1;
  g1[0] = 0x00010000;          // data_size=1 (2B); mask/pad/iterate = 0
  g1[1] = (int)(256u << 16);   // tensor_dim0[15:0] in bits 63:48
  g1[2] = (int)(256u << 16);   // tensor_dim0 hi=0 | tensor_dim1[15:0]=256 in bits 95:80
  g1[3] = (int)(32u << 16);    // tensor_dim1 hi=0 | tile_dim0=32 in bits 127:112
  g1[4] = 256;                 // tile_dim1=256 | tile_dim2=0
  g1[5] = 256;                 // tensor_dim0_stride[31:0]=256
  g1[6] = 0;                   // stride hi | tensor_dim1_stride lo
  g1[7] = 0;
  i32x4 z4 = {0, 0, 0, 0};
#if defined(__clang_major__) && __clang_major__ >= 23
  i32x8 z8 = {0, 0, 0, 0, 0, 0, 0, 0};
  __builtin_amdgcn_tensor_load_to_lds(g0, g1, z4, z4, z8, 0);
#else
  __builtin_amdgcn_tensor_load_to_lds(g0, g1, z4, z4, 0);
#endif
}
#define GEMM_AW 1
#else
#define GEMM_AW 5
#endif

static __device__ __forceinline__ float wave_sum(float v) {
#pragma unroll
  for (int o = 16; o > 0; o >>= 1) v += __shfl_xor(v, o, 32);
  return v;
}
static __device__ __forceinline__ float wave_max(float v) {
#pragma unroll
  for (int o = 16; o > 0; o >>= 1) v = fmaxf(v, __shfl_xor(v, o, 32));
  return v;
}

// ---------------- packing kernels ----------------

__global__ void pack_conv_w_kernel(const float* __restrict__ w, unsigned short* __restrict__ wp) {
  int idx = blockIdx.x * 256 + threadIdx.x;
  if (idx >= 3 * C_ * C_) return;
  int o = idx / (3 * C_);
  int rem = idx - o * (3 * C_);
  int i = rem / 3;
  int j = rem - i * 3;
  wp[(size_t)j * C_ * C_ + (size_t)o * C_ + i] = f2bf(w[idx]);
}

__global__ void pack_lin_w_kernel(const float* __restrict__ w, unsigned short* __restrict__ wp) {
  int idx = blockIdx.x * 256 + threadIdx.x;
  if (idx >= C_ * C_) return;
  wp[idx] = f2bf(w[idx]);
}

__global__ void pack_x_kernel(const float* __restrict__ x, unsigned short* __restrict__ xpad) {
  size_t idx = (size_t)blockIdx.x * 256 + threadIdx.x;
  const size_t total = (size_t)B_ * (T_ + 2) * C_;
  if (idx >= total) return;
  size_t b = idx / ((size_t)(T_ + 2) * C_);
  size_t rem = idx - b * ((size_t)(T_ + 2) * C_);
  int t = (int)(rem / C_);
  int c = (int)(rem & (C_ - 1));
  float v = 0.f;
  if (t >= 1 && t <= T_) v = x[(b * T_ + (t - 1)) * C_ + c];
  xpad[idx] = f2bf(v);
}

// ---------------- generic WMMA GEMM: 32 rows x 256 cols per block ----------------
// Double-buffered async/TDM pipeline. 8 waves = 2 row-groups x 4 col-groups.
__global__ void gemm32_kernel(const unsigned short* __restrict__ Abase, size_t aBatch,
                              const unsigned short* __restrict__ Wp,
                              const float* __restrict__ bias,
                              int shifts, int doRelu,
                              unsigned short* __restrict__ outBf,
                              unsigned short* __restrict__ outBfT,
                              float* __restrict__ outF) {
  extern __shared__ char smem[];
  unsigned short* As = (unsigned short*)smem;             // [2][32][32]
  unsigned short* Ws = (unsigned short*)(smem + 4096);    // [2][256][32]
  const int tid = threadIdx.x;
  const int wave = tid >> 5, lane = tid & 31;
  const int rg = wave >> 2, cg = wave & 3;
  const int b = blockIdx.x >> 5;
  const int m0 = (blockIdx.x & 31) << 5;
  const int mrow = lane & 15, kh = lane >> 4;
  const int ncol = lane & 15, mbase = kh << 3;
  const int n0 = cg << 6;
  const unsigned short* Ab = Abase + (size_t)b * aBatch;
  const int nsteps = shifts << 3;
  const int am = tid >> 3, aq = tid & 7;   // A tile: 256 lanes x 8B
  v8f acc[4] = {};

  auto stage = [&](int step, int buf) {
    const int j = step >> 3, k0 = (step & 7) << 5;
    async_b64(As + buf * 1024 + tid * 4,
              Ab + (size_t)(m0 + am + j) * C_ + k0 + aq * 4);
#if HAVE_TDM
    if (wave == 0)
      tdm_load_2d((unsigned)(size_t)Ws + (unsigned)buf * 16384u,
                  Wp + (size_t)j * C_ * C_ + k0);
#else
    const unsigned short* Wj = Wp + (size_t)j * C_ * C_;
#pragma unroll
    for (int u = 0; u < 4; ++u) {
      int i = tid + u * 256;
      int r = i >> 2, q4 = i & 3;
      async_b128(Ws + buf * 8192 + i * 8, Wj + (size_t)r * C_ + k0 + q4 * 8);
    }
#endif
  };

  stage(0, 0);
  for (int step = 0; step < nsteps; ++step) {
    const int cur = step & 1;
    if (step + 1 < nsteps) {
      stage(step + 1, cur ^ 1);
      async_waitN<GEMM_AW>();
#if HAVE_TDM
      if (wave == 0) tensor_waitN<1>();
#endif
    } else {
      async_waitN<0>();
#if HAVE_TDM
      if (wave == 0) tensor_waitN<0>();
#endif
    }
    __syncthreads();
    u16x16 af = ld_afrag(As + cur * 1024 + rg * 512, mrow, kh);
#pragma unroll
    for (int q = 0; q < 4; ++q) {
      u16x16 bf = ld_bfrag(Ws + cur * 8192, n0 + q * 16 + ncol, kh);
      acc[q] = wmma_bf16(af, bf, acc[q]);
    }
    __syncthreads();
  }
#pragma unroll
  for (int q = 0; q < 4; ++q) {
    int n = n0 + q * 16 + ncol;
    float bs = bias[n];
#pragma unroll
    for (int r = 0; r < 8; ++r) {
      int m = m0 + rg * 16 + mbase + r;
      float v = acc[q][r] + bs;
      if (doRelu) v = fmaxf(v, 0.f);
      size_t base = ((size_t)b * T_ + m) * C_ + n;
      if (outBf)  outBf[base] = f2bf(v);
      if (outBfT) outBfT[((size_t)b * C_ + n) * T_ + m] = f2bf(v);
      if (outF)   outF[base] = v;
    }
  }
}

// ---------------- attention scores: S = (Q K^T)/16 ----------------
// grid = (B_*T_/32, 4); 32 query rows x 256 key cols per block; double-buffered async.
__global__ void scores_kernel(const unsigned short* __restrict__ q,
                              const unsigned short* __restrict__ kmat,
                              float* __restrict__ S) {
  extern __shared__ char smem[];
  unsigned short* As = (unsigned short*)smem;             // [2][32][32]
  unsigned short* Ks = (unsigned short*)(smem + 4096);    // [2][256][32]
  const int tid = threadIdx.x, wave = tid >> 5, lane = tid & 31;
  const int rg = wave >> 2, cg = wave & 3;
  const int b = blockIdx.x >> 5, m0 = (blockIdx.x & 31) << 5;
  const int cb = blockIdx.y;
  const int mrow = lane & 15, kh = lane >> 4, ncol = lane & 15, mbase = kh << 3;
  const int n0 = cg << 6;
  const unsigned short* qb = q + ((size_t)b * T_ + m0) * C_;
  const unsigned short* kb = kmat + ((size_t)b * T_ + cb * 256) * C_;
  const int am = tid >> 3, aq = tid & 7;
  v8f acc[4] = {};

  auto stage = [&](int step, int buf) {
    const int k0 = step << 5;
    async_b64(As + buf * 1024 + tid * 4, qb + (size_t)am * C_ + k0 + aq * 4);
#pragma unroll
    for (int u = 0; u < 4; ++u) {
      int i = tid + u * 256;
      int r = i >> 2, q4 = i & 3;
      async_b128(Ks + buf * 8192 + i * 8, kb + (size_t)r * C_ + k0 + q4 * 8);
    }
  };

  stage(0, 0);
  for (int step = 0; step < 8; ++step) {
    const int cur = step & 1;
    if (step + 1 < 8) { stage(step + 1, cur ^ 1); async_waitN<5>(); }
    else              { async_waitN<0>(); }
    __syncthreads();
    u16x16 af = ld_afrag(As + cur * 1024 + rg * 512, mrow, kh);
#pragma unroll
    for (int qi = 0; qi < 4; ++qi) {
      u16x16 bf = ld_bfrag(Ks + cur * 8192, n0 + qi * 16 + ncol, kh);
      acc[qi] = wmma_bf16(af, bf, acc[qi]);
    }
    __syncthreads();
  }
  const float scale = 0.0625f;  // 1/sqrt(256)
  float* Sb = S + (size_t)b * T_ * T_;
#pragma unroll
  for (int qi = 0; qi < 4; ++qi) {
    int n = cb * 256 + n0 + qi * 16 + ncol;
#pragma unroll
    for (int r = 0; r < 8; ++r) {
      int m = m0 + rg * 16 + mbase + r;
      Sb[(size_t)m * T_ + n] = acc[qi][r] * scale;
    }
  }
}

// ---------------- softmax (in place) + score_his ----------------
__global__ void softmax_kernel(float* __restrict__ S, const float* __restrict__ shis) {
  const int row = blockIdx.x * 8 + (threadIdx.x >> 5);
  const int lane = threadIdx.x & 31;
  float* p = S + (size_t)row * T_;
  const float* hp = shis + (size_t)row * T_;
  float v[32];
  float mx = -3.4e38f;
#pragma unroll
  for (int i = 0; i < 32; ++i) { v[i] = p[lane + i * 32]; mx = fmaxf(mx, v[i]); }
  mx = wave_max(mx);
  float sum = 0.f;
#pragma unroll
  for (int i = 0; i < 32; ++i) { v[i] = __expf(v[i] - mx); sum += v[i]; }
  sum = wave_sum(sum);
  const float inv = 1.f / sum;
#pragma unroll
  for (int i = 0; i < 32; ++i) p[lane + i * 32] = v[i] * inv + hp[lane + i * 32];
}

// ---------------- AV: out = A @ V + x ----------------
// A fp32 -> bf16 via VALU staging; V async double-buffered. 32 rows x 256 cols per block.
__global__ void av_kernel(const float* __restrict__ A, const unsigned short* __restrict__ vT,
                          const float* __restrict__ x, float* __restrict__ out) {
  extern __shared__ char smem[];
  unsigned short* As = (unsigned short*)smem;             // [2][32][32]
  unsigned short* Vs = (unsigned short*)(smem + 4096);    // [2][256][32]
  const int tid = threadIdx.x, wave = tid >> 5, lane = tid & 31;
  const int rg = wave >> 2, cg = wave & 3;
  const int b = blockIdx.x >> 5, m0 = (blockIdx.x & 31) << 5;
  const int mrow = lane & 15, kh = lane >> 4, ncol = lane & 15, mbase = kh << 3;
  const int n0 = cg << 6;
  const float* Ab = A + ((size_t)b * T_ + m0) * T_;
  const unsigned short* vb = vT + (size_t)b * C_ * T_;
  v8f acc[4] = {};

  auto stage = [&](int step, int buf) {
    const int k0 = step << 5;
#pragma unroll
    for (int u = 0; u < 4; ++u) {  // V tile: 1024 x 16B async
      int i = tid + u * 256;
      int r = i >> 2, q4 = i & 3;
      async_b128(Vs + buf * 8192 + i * 8, vb + (size_t)r * T_ + k0 + q4 * 8);
    }
#pragma unroll
    for (int u = 0; u < 4; ++u) {  // A tile f32->bf16 via VALU
      int i = tid + u * 256;
      int m = i >> 5, kk = i & 31;
      As[buf * 1024 + i] = f2bf(Ab[(size_t)m * T_ + k0 + kk]);
    }
  };

  stage(0, 0);
  for (int step = 0; step < 32; ++step) {
    const int cur = step & 1;
    if (step + 1 < 32) { stage(step + 1, cur ^ 1); async_waitN<4>(); }
    else               { async_waitN<0>(); }
    __syncthreads();
    u16x16 af = ld_afrag(As + cur * 1024 + rg * 512, mrow, kh);
#pragma unroll
    for (int q = 0; q < 4; ++q) {
      u16x16 bf = ld_bfrag(Vs + cur * 8192, n0 + q * 16 + ncol, kh);
      acc[q] = wmma_bf16(af, bf, acc[q]);
    }
    __syncthreads();
  }
#pragma unroll
  for (int q = 0; q < 4; ++q) {
    int n = n0 + q * 16 + ncol;
#pragma unroll
    for (int r = 0; r < 8; ++r) {
      int m = m0 + rg * 16 + mbase + r;
      size_t base = ((size_t)b * T_ + m) * C_ + n;
      out[base] = acc[q][r] + x[base];
    }
  }
}

// ---------------- layernorm (one wave per 256-elem row) ----------------
__global__ void ln_kernel(const float* __restrict__ in, const float* __restrict__ add,
                          const float* __restrict__ g, const float* __restrict__ bb,
                          float* __restrict__ outF, unsigned short* __restrict__ outBf) {
  const int row = blockIdx.x * 8 + (threadIdx.x >> 5);
  const int lane = threadIdx.x & 31;
  const size_t base = (size_t)row * C_;
  float v[8];
  float s = 0.f;
#pragma unroll
  for (int i = 0; i < 8; ++i) {
    int c = lane + i * 32;
    v[i] = in[base + c];
    if (add) v[i] += add[base + c];
    s += v[i];
  }
  s = wave_sum(s);
  const float mean = s * (1.f / C_);
  float var = 0.f;
#pragma unroll
  for (int i = 0; i < 8; ++i) { float d = v[i] - mean; var += d * d; }
  var = wave_sum(var) * (1.f / C_);
  const float rstd = rsqrtf(var + 1e-5f);
#pragma unroll
  for (int i = 0; i < 8; ++i) {
    int c = lane + i * 32;
    float o = (v[i] - mean) * rstd * g[c] + bb[c];
    if (outF)  outF[base + c] = o;
    if (outBf) outBf[base + c] = f2bf(o);
  }
}

// ---------------- launch ----------------

extern "C" void kernel_launch(void* const* d_in, const int* in_sizes, int n_in,
                              void* d_out, int out_size, void* d_ws, size_t ws_size,
                              hipStream_t stream) {
  (void)in_sizes; (void)n_in; (void)out_size; (void)ws_size;
  const float* x      = (const float*)d_in[0];
  const float* shis   = (const float*)d_in[1];
  const float* conv1w = (const float*)d_in[2];
  const float* conv1b = (const float*)d_in[3];
  const float* conv2w = (const float*)d_in[4];
  const float* conv2b = (const float*)d_in[5];
  const float* vffw   = (const float*)d_in[6];
  const float* vffb   = (const float*)d_in[7];
  const float* lng    = (const float*)d_in[8];
  const float* lnb    = (const float*)d_in[9];
  const float* ffw1   = (const float*)d_in[10];
  const float* ffb1   = (const float*)d_in[11];
  const float* ffw2   = (const float*)d_in[12];
  const float* ffb2   = (const float*)d_in[13];
  const float* lnffg  = (const float*)d_in[14];
  const float* lnffb  = (const float*)d_in[15];

  float* out  = (float*)d_out;
  float* Sreg = out + (size_t)B_ * T_ * C_;   // new_score_his region (also A scratch)

  char* w = (char*)d_ws;
  size_t off = 0;
  auto alloc = [&](size_t bytes) -> void* {
    void* p = w + off;
    off += (bytes + 255) & ~(size_t)255;
    return p;
  };
  unsigned short* w1p   = (unsigned short*)alloc((size_t)3 * C_ * C_ * 2);
  unsigned short* w2p   = (unsigned short*)alloc((size_t)3 * C_ * C_ * 2);
  unsigned short* vffp  = (unsigned short*)alloc((size_t)C_ * C_ * 2);
  unsigned short* ff1p  = (unsigned short*)alloc((size_t)C_ * C_ * 2);
  unsigned short* ff2p  = (unsigned short*)alloc((size_t)C_ * C_ * 2);
  unsigned short* xpad  = (unsigned short*)alloc((size_t)B_ * (T_ + 2) * C_ * 2);
  unsigned short* qbf   = (unsigned short*)alloc((size_t)B_ * T_ * C_ * 2);
  unsigned short* kbf   = (unsigned short*)alloc((size_t)B_ * T_ * C_ * 2);
  unsigned short* vbfT  = (unsigned short*)alloc((size_t)B_ * T_ * C_ * 2);
  unsigned short* vlnbf = (unsigned short*)alloc((size_t)B_ * T_ * C_ * 2);
  unsigned short* hbf   = (unsigned short*)alloc((size_t)B_ * T_ * C_ * 2);
  float* attnx = (float*)alloc((size_t)B_ * T_ * C_ * 4);
  float* vln   = (float*)alloc((size_t)B_ * T_ * C_ * 4);
  float* h2f   = (float*)alloc((size_t)B_ * T_ * C_ * 4);

  const int smemG = 4096 + 2 * C_ * 32 * 2;  // 36864 bytes: 2xA tile + 2xB tile

  pack_conv_w_kernel<<<768, 256, 0, stream>>>(conv1w, w1p);
  pack_conv_w_kernel<<<768, 256, 0, stream>>>(conv2w, w2p);
  pack_lin_w_kernel<<<256, 256, 0, stream>>>(vffw, vffp);
  pack_lin_w_kernel<<<256, 256, 0, stream>>>(ffw1, ff1p);
  pack_lin_w_kernel<<<256, 256, 0, stream>>>(ffw2, ff2p);
  pack_x_kernel<<<B_ * (T_ + 2), 256, 0, stream>>>(x, xpad);

  const size_t padBatch = (size_t)(T_ + 2) * C_;
  const int gemmGrid = B_ * T_ / 32;
  gemm32_kernel<<<gemmGrid, 256, smemG, stream>>>(xpad, padBatch, w1p, conv1b, 3, 0, qbf, nullptr, nullptr);
  gemm32_kernel<<<gemmGrid, 256, smemG, stream>>>(xpad, padBatch, w2p, conv2b, 3, 0, kbf, nullptr, nullptr);
  gemm32_kernel<<<gemmGrid, 256, smemG, stream>>>(xpad + C_, padBatch, vffp, vffb, 1, 0, nullptr, vbfT, nullptr);

  scores_kernel<<<dim3(gemmGrid, 4), 256, smemG, stream>>>(qbf, kbf, Sreg);
  softmax_kernel<<<B_ * T_ / 8, 256, 0, stream>>>(Sreg, shis);
  av_kernel<<<gemmGrid, 256, smemG, stream>>>(Sreg, vbfT, x, attnx);
  ln_kernel<<<B_ * T_ / 8, 256, 0, stream>>>(attnx, nullptr, lng, lnb, vln, vlnbf);
  gemm32_kernel<<<gemmGrid, 256, smemG, stream>>>(vlnbf, (size_t)T_ * C_, ff1p, ffb1, 1, 1, hbf, nullptr, nullptr);
  gemm32_kernel<<<gemmGrid, 256, smemG, stream>>>(hbf, (size_t)T_ * C_, ff2p, ffb2, 1, 0, nullptr, nullptr, h2f);
  ln_kernel<<<B_ * T_ / 8, 256, 0, stream>>>(h2f, vln, lnffg, lnffb, out, nullptr);
}
